// PatchedQwen3MoeExperts_55714315763865
// MI455X (gfx1250) — matrix operand order repaired and checked by reading, not compiled, not run on hardware
//
#include <hip/hip_runtime.h>
#include <stdint.h>

#define T     8192
#define H     2048
#define IDIM  768
#define NE    16
#define TOPK  2
#define CAP   (T*TOPK)        // per-expert bucket capacity upper bound
#define MT    32              // tokens per block tile
#define TILES_PER_E (CAP/MT)  // 512

typedef __attribute__((ext_vector_type(16))) __bf16 v16bf;
typedef __attribute__((ext_vector_type(8)))  __bf16 v8bf;
typedef __attribute__((ext_vector_type(8)))  float  v8f;

#define CAT16(lo, hi) __builtin_shufflevector((lo), (hi), 0,1,2,3,4,5,6,7,8,9,10,11,12,13,14,15)

static __device__ __forceinline__ uint16_t f2bfu(float f) {
  // round-to-nearest-even fp32 -> bf16
  uint32_t u = __builtin_bit_cast(uint32_t, f);
  uint32_t r = u + 0x7FFFu + ((u >> 16) & 1u);
  return (uint16_t)(r >> 16);
}

// ---------------- fp32 -> bf16 bulk convert (vec4) ----------------
__global__ void k_cvt_bf16(const float* __restrict__ src, uint16_t* __restrict__ dst, int n4) {
  int i = blockIdx.x * blockDim.x + threadIdx.x;
  if (i >= n4) return;
  float4 f = ((const float4*)src)[i];
  uint2 o;
  o.x = (uint32_t)f2bfu(f.x) | ((uint32_t)f2bfu(f.y) << 16);
  o.y = (uint32_t)f2bfu(f.z) | ((uint32_t)f2bfu(f.w) << 16);
  ((uint2*)dst)[i] = o;
}

// ---------------- routing: scatter (token, weight) into expert buckets ----------------
__global__ void k_route(const int* __restrict__ tki, const float* __restrict__ tkw,
                        int* __restrict__ cnt, int* __restrict__ bidx, float* __restrict__ bw) {
  int i = blockIdx.x * blockDim.x + threadIdx.x;
  if (i >= T * TOPK) return;
  int t = i >> 1;
  int e = tki[i];
  float w = tkw[i];
  int pos = atomicAdd(&cnt[e], 1);
  bidx[e * CAP + pos] = t;
  bw[e * CAP + pos]   = w;
}

// ---------------- fused grouped-GEMM MoE expert kernel ----------------
__launch_bounds__(256)
__global__ void k_moe(const uint16_t* __restrict__ hsb,   // [T][H] bf16
                      const uint16_t* __restrict__ w1b,   // [E][2I][H] bf16
                      const uint16_t* __restrict__ w2b,   // [E][H][I] bf16
                      const int* __restrict__ cnt,
                      const int* __restrict__ bidx,
                      const float* __restrict__ bw,
                      float* __restrict__ out) {
  __shared__ uint16_t act[MT * IDIM];   // 48 KB bf16 activations
  __shared__ int   toks[MT];
  __shared__ float wrow[MT];

  const int e    = blockIdx.x / TILES_PER_E;
  const int tile = blockIdx.x % TILES_PER_E;
  const int base = tile * MT;
  const int ce   = cnt[e];
  if (base >= ce) return;          // uniform per-block exit

  const int tid = threadIdx.x;
  if (tid < MT) {
    int r = base + tid;
    bool v = r < ce;
    toks[tid] = bidx[e * CAP + (v ? r : base)];  // pad with a valid token, weight 0
    wrow[tid] = v ? bw[e * CAP + r] : 0.0f;
  }
  __syncthreads();

  const int wid  = tid >> 5;
  const int lane = tid & 31;
  const int lrow = lane & 15;
  const int hf   = lane >> 4;

  const int tok0 = toks[lrow];
  const int tok1 = toks[16 + lrow];
  const uint16_t* a0p = hsb + (size_t)tok0 * H;
  const uint16_t* a1p = hsb + (size_t)tok1 * H;

  // ---- Phase A: act[MT x I] = silu(hs @ Wg^T) * (hs @ Wu^T) ----
  for (int nt = wid; nt < IDIM / 16; nt += 8) {
    const int n0 = nt * 16;
    const uint16_t* bg = w1b + ((size_t)e * 2 * IDIM + n0 + lrow) * H + hf * 8;
    const uint16_t* bu = w1b + ((size_t)e * 2 * IDIM + IDIM + n0 + lrow) * H + hf * 8;

    v8f cg0 = {}, cg1 = {}, cu0 = {}, cu1 = {};
    for (int k0 = 0; k0 < H; k0 += 32) {
      v16bf A0 = CAT16(*(const v8bf*)(a0p + k0 + hf * 8),
                       *(const v8bf*)(a0p + k0 + 16 + hf * 8));
      v16bf A1 = CAT16(*(const v8bf*)(a1p + k0 + hf * 8),
                       *(const v8bf*)(a1p + k0 + 16 + hf * 8));
      v16bf Bg = CAT16(*(const v8bf*)(bg + k0), *(const v8bf*)(bg + k0 + 16));
      v16bf Bu = CAT16(*(const v8bf*)(bu + k0), *(const v8bf*)(bu + k0 + 16));
      cg0 = __builtin_amdgcn_wmma_f32_16x16x32_bf16(false, A0, false, Bg, (short)0, cg0, false, false);
      cg1 = __builtin_amdgcn_wmma_f32_16x16x32_bf16(false, A1, false, Bg, (short)0, cg1, false, false);
      cu0 = __builtin_amdgcn_wmma_f32_16x16x32_bf16(false, A0, false, Bu, (short)0, cu0, false, false);
      cu1 = __builtin_amdgcn_wmma_f32_16x16x32_bf16(false, A1, false, Bu, (short)0, cu1, false, false);
    }
    const int n = n0 + lrow;
#pragma unroll
    for (int r = 0; r < 8; ++r) {
      {
        int m = r + 8 * hf;
        float g = cg0[r], u = cu0[r];
        float s = g / (1.0f + __expf(-g));
        act[m * IDIM + n] = f2bfu(s * u);
      }
      {
        int m = 16 + r + 8 * hf;
        float g = cg1[r], u = cu1[r];
        float s = g / (1.0f + __expf(-g));
        act[m * IDIM + n] = f2bfu(s * u);
      }
    }
  }
  __syncthreads();

  // per-lane output row weights / tokens (C layout: elem r -> m = r + 8*hf)
  float wv0[8], wv1[8];
  int   tv0[8], tv1[8];
#pragma unroll
  for (int r = 0; r < 8; ++r) {
    int m0 = r + 8 * hf;
    int m1 = 16 + r + 8 * hf;
    wv0[r] = wrow[m0]; tv0[r] = toks[m0];
    wv1[r] = wrow[m1]; tv1[r] = toks[m1];
  }

  // ---- Phase B: out += combine * (act @ W2^T) ----
  for (int nt = wid; nt < H / 16; nt += 8) {
    const int n0 = nt * 16;
    const uint16_t* b2 = w2b + ((size_t)e * H + n0 + lrow) * IDIM + hf * 8;
    v8f c0 = {}, c1 = {};
    for (int k0 = 0; k0 < IDIM; k0 += 32) {
      v16bf A0 = CAT16(*(const v8bf*)&act[lrow * IDIM + k0 + hf * 8],
                       *(const v8bf*)&act[lrow * IDIM + k0 + 16 + hf * 8]);
      v16bf A1 = CAT16(*(const v8bf*)&act[(16 + lrow) * IDIM + k0 + hf * 8],
                       *(const v8bf*)&act[(16 + lrow) * IDIM + k0 + 16 + hf * 8]);
      v16bf B  = CAT16(*(const v8bf*)(b2 + k0), *(const v8bf*)(b2 + k0 + 16));
      c0 = __builtin_amdgcn_wmma_f32_16x16x32_bf16(false, A0, false, B, (short)0, c0, false, false);
      c1 = __builtin_amdgcn_wmma_f32_16x16x32_bf16(false, A1, false, B, (short)0, c1, false, false);
    }
    const int col = n0 + lrow;
#pragma unroll
    for (int r = 0; r < 8; ++r) {
      if (wv0[r] != 0.0f) atomicAdd(out + (size_t)tv0[r] * H + col, wv0[r] * c0[r]);
      if (wv1[r] != 0.0f) atomicAdd(out + (size_t)tv1[r] * H + col, wv1[r] * c1[r]);
    }
  }
}

extern "C" void kernel_launch(void* const* d_in, const int* in_sizes, int n_in,
                              void* d_out, int out_size, void* d_ws, size_t ws_size,
                              hipStream_t stream) {
  (void)in_sizes; (void)n_in; (void)out_size; (void)ws_size;
  const float* hs  = (const float*)d_in[0];
  const int*   tki = (const int*)d_in[1];
  const float* tkw = (const float*)d_in[2];
  const float* gup = (const float*)d_in[3];
  const float* dwn = (const float*)d_in[4];
  float* out = (float*)d_out;

  char* p = (char*)d_ws;
  int*      cnt  = (int*)p;       p += 256;
  int*      bidx = (int*)p;       p += (size_t)NE * CAP * 4;
  float*    bw   = (float*)p;     p += (size_t)NE * CAP * 4;
  uint16_t* hsb  = (uint16_t*)p;  p += (size_t)T * H * 2;
  uint16_t* w1b  = (uint16_t*)p;  p += (size_t)NE * 2 * IDIM * H * 2;
  uint16_t* w2b  = (uint16_t*)p;  p += (size_t)NE * H * IDIM * 2;

  hipMemsetAsync(out, 0, (size_t)T * H * 4, stream);
  hipMemsetAsync(cnt, 0, 256, stream);

  int n4 = T * H / 4;
  k_cvt_bf16<<<(n4 + 255) / 256, 256, 0, stream>>>(hs, hsb, n4);
  n4 = NE * 2 * IDIM * H / 4;
  k_cvt_bf16<<<(n4 + 255) / 256, 256, 0, stream>>>(gup, w1b, n4);
  n4 = NE * H * IDIM / 4;
  k_cvt_bf16<<<(n4 + 255) / 256, 256, 0, stream>>>(dwn, w2b, n4);

  k_route<<<(T * TOPK + 255) / 256, 256, 0, stream>>>(tki, tkw, cnt, bidx, bw);

  k_moe<<<NE * TILES_PER_E, 256, 0, stream>>>(hsb, w1b, w2b, cnt, bidx, bw, out);
}
